// MetabolicBrain_84808424227250
// MI455X (gfx1250) — compile-verified, matrix operand order checked by reading, not run on hardware
//
#include <hip/hip_runtime.h>

// ---------------------------------------------------------------------------
// MetabolicBrain for MI455X (gfx1250, wave32, WMMA bf16 16x16x32)
// B=12 T=8192 D=260 V=256 M=280 DEPTH=8
// ---------------------------------------------------------------------------

#define BB 12
#define TT 8192
#define DD 260
#define VV 256
#define MM 280
#define G3 780          // 3*D
#define DEPTH_C 8
#define KPAD 288        // D padded to 9*32 for K-tiling
#define KT9 9           // K tiles of 32
#define NT49 49         // ceil(780/16)
#define NT16 16         // 256/16
#define BD 3120         // B*D
#define S_BIAS_F 4.6051702f            // ln(100)
#define COST_STEP (0.12f * 8.0f / 8192.0f)

typedef __bf16 bf16;
typedef __attribute__((ext_vector_type(16))) __bf16 v16bf;
typedef __attribute__((ext_vector_type(8)))  float  v8f;

// K index inside a 16x32 bf16 A/B fragment for (element e, lane-group g).
__device__ __forceinline__ int ak_map(int e, int g) {
  return (e < 8) ? (g * 8 + e) : (16 + g * 8 + (e - 8));
}

__device__ __forceinline__ v8f wmma_bf16(v16bf a, v16bf b, v8f c) {
  return __builtin_amdgcn_wmma_f32_16x16x32_bf16(
      /*neg_a=*/false, a, /*neg_b=*/false, b,
      /*c_mod=*/(short)0, c, /*reuse_a=*/false, /*reuse_b=*/false);
}

// Build an A fragment (16xK tile, k-tile kt) from an f32 LDS matrix.
__device__ __forceinline__ v16bf afrag_f32(const float* base, int ld, int rows,
                                           int kt, int lane) {
  int m = lane & 15, g = lane >> 4;
  v16bf a;
#pragma unroll
  for (int e = 0; e < 16; ++e) {
    int k = kt * 32 + ak_map(e, g);
    float v = (m < rows && k < DD) ? base[m * ld + k] : 0.f;
    a[e] = (bf16)v;
  }
  return a;
}

// Build an A fragment from a bf16 LDS matrix.
__device__ __forceinline__ v16bf afrag_bf16(const bf16* base, int ld, int rows,
                                            int kt, int lane) {
  int m = lane & 15, g = lane >> 4;
  v16bf a;
#pragma unroll
  for (int e = 0; e < 16; ++e) {
    int k = kt * 32 + ak_map(e, g);
    a[e] = (m < rows && k < DD) ? base[m * ld + k] : (bf16)0.f;
  }
  return a;
}

// B fragments are pre-swizzled in workspace: [ntile][ktile][lane][16 bf16].
__device__ __forceinline__ v16bf bfrag(const bf16* w, int nt, int kt, int lane) {
  return reinterpret_cast<const v16bf*>(w)[(nt * KT9 + kt) * 32 + lane];
}

__device__ __forceinline__ float sigf(float x) { return 1.f / (1.f + expf(-x)); }

// ---------------------------------------------------------------------------
// Prep: l2-normalized memory keys (f32) + soma row norms.
// ---------------------------------------------------------------------------
__global__ void prep_norms_kernel(const float* __restrict__ mem_keys,
                                  const float* __restrict__ soma_w,
                                  float* __restrict__ keysn,
                                  float* __restrict__ somanorm) {
  int tid = threadIdx.x;
  if (tid < MM) {
    float s = 0.f;
    for (int k = 0; k < DD; ++k) { float v = mem_keys[tid * DD + k]; s += v * v; }
    float inv = 1.f / fmaxf(sqrtf(s), 1e-12f);
    for (int k = 0; k < DD; ++k) keysn[tid * DD + k] = mem_keys[tid * DD + k] * inv;
  } else if (tid < MM + VV) {
    int r = tid - MM;
    float s = 0.f;
    for (int k = 0; k < DD; ++k) { float v = soma_w[r * DD + k]; s += v * v; }
    somanorm[r] = fmaxf(sqrtf(s), 1e-12f);
  }
}

// ---------------------------------------------------------------------------
// Convert an (N x K) f32 weight into pre-swizzled bf16 WMMA B fragments.
// Optional per-row normalization (for l2n(soma_w)).
// ---------------------------------------------------------------------------
__global__ void swizzle_kernel(const float* __restrict__ W,
                               const float* __restrict__ rownorm,
                               int N, int K, int NT, bf16* __restrict__ dst) {
  int f = blockIdx.x * blockDim.x + threadIdx.x;
  int total = NT * KT9 * 512;
  if (f >= total) return;
  int e = f & 15;
  int lane = (f >> 4) & 31;
  int tile = f >> 9;
  int kt = tile % KT9;
  int nt = tile / KT9;
  int n = nt * 16 + (lane & 15);
  int k = kt * 32 + ak_map(e, lane >> 4);
  float v = 0.f;
  if (n < N && k < K) {
    v = W[n * K + k];
    if (rownorm) v /= rownorm[n];
  }
  dst[f] = (bf16)v;
}

// ---------------------------------------------------------------------------
// gi[b,t,:] = soma_w[x[b,t]] @ gru_wih^T + gru_bih   (M=B*T, N=780, K=260)
// One block per 16-row M tile; 8 waves x 49 N tiles; WMMA bf16.
// ---------------------------------------------------------------------------
__global__ __launch_bounds__(256) void gi_kernel(const int* __restrict__ x,
                                                 const float* __restrict__ soma_w,
                                                 const bf16* __restrict__ wih_bf,
                                                 const float* __restrict__ bih,
                                                 float* __restrict__ gi) {
  __shared__ bf16 As[16 * KPAD];
  int tid = threadIdx.x;
  int m0 = blockIdx.x * 16;
  for (int s = tid; s < 16 * KPAD; s += 256) {
    int r = s / KPAD, k = s - r * KPAD;
    float v = 0.f;
    if (k < DD) { int xv = x[m0 + r]; v = soma_w[(size_t)xv * DD + k]; }
    As[s] = (bf16)v;
  }
  __syncthreads();
  int wv = tid >> 5, lane = tid & 31;
  int g = lane >> 4, nlane = lane & 15;
  for (int nt = wv; nt < NT49; nt += 8) {
    v8f c = {};
#pragma unroll
    for (int kt = 0; kt < KT9; ++kt)
      c = wmma_bf16(afrag_bf16(As, KPAD, 16, kt, lane), bfrag(wih_bf, nt, kt, lane), c);
    int n = nt * 16 + nlane;
    if (n < G3) {
#pragma unroll
      for (int r = 0; r < 8; ++r) {
        int m = m0 + g * 8 + r;
        gi[(size_t)m * G3 + n] = c[r] + bih[n];
      }
    }
  }
}

// ---------------------------------------------------------------------------
// Forward GRU scan over T. Persistent single workgroup (32 waves),
// hidden state resident in LDS, per-step gh via WMMA.
// ---------------------------------------------------------------------------
__global__ __launch_bounds__(1024) void fgru_kernel(const float* __restrict__ gi,
                                                    const bf16* __restrict__ whh_bf,
                                                    const float* __restrict__ bhh,
                                                    const float* __restrict__ h0,
                                                    float* __restrict__ pf,
                                                    float* __restrict__ hf_out) {
  __shared__ float hbuf[BD];        // (12,260) fp32 state
  __shared__ float ghb[BB * G3];    // (12,780) gate preacts
  int tid = threadIdx.x;
  int wv = tid >> 5, lane = tid & 31;
  int g = lane >> 4, nlane = lane & 15;
  for (int i = tid; i < BD; i += 1024) hbuf[i] = h0[i];
  __syncthreads();
  for (int t = 0; t < TT; ++t) {
    // gh = h @ whh^T + bhh : 49 N tiles over 32 waves, 9 WMMA k-steps each
    for (int nt = wv; nt < NT49; nt += 32) {
      v8f c = {};
#pragma unroll
      for (int kt = 0; kt < KT9; ++kt)
        c = wmma_bf16(afrag_f32(hbuf, DD, BB, kt, lane), bfrag(whh_bf, nt, kt, lane), c);
      int n = nt * 16 + nlane;
      if (n < G3) {
#pragma unroll
        for (int r = 0; r < 8; ++r) {
          int m = g * 8 + r;
          if (m < BB) ghb[m * G3 + n] = c[r] + bhh[n];
        }
      }
    }
    __syncthreads();
    for (int idx = tid; idx < BD; idx += 1024) {
      int b = idx / DD, d = idx - b * DD;
      const float* gir = gi + ((size_t)b * TT + t) * G3;   // bih already folded in
      float rr = sigf(gir[d] + ghb[b * G3 + d]);
      float zz = sigf(gir[DD + d] + ghb[b * G3 + DD + d]);
      float nn = tanhf(gir[2 * DD + d] + rr * ghb[b * G3 + 2 * DD + d]);
      float hnew = (1.f - zz) * nn + zz * hbuf[idx];
      hbuf[idx] = hnew;
      pf[(size_t)t * BD + idx] = hnew;
      if (t + 1 < TT) __builtin_prefetch(gir + G3, 0, 0);  // next-step gi row
    }
    __syncthreads();
  }
  for (int i = tid; i < BD; i += 1024) hf_out[i] = hbuf[i];
}

// ---------------------------------------------------------------------------
// tstep scan: DEPTH=8 inner GRU cells (WMMA), memory attention, pfc+LN+gelu.
// Single persistent workgroup; all per-step state in LDS (~62.5 KB).
// ---------------------------------------------------------------------------
__global__ __launch_bounds__(1024) void tstep_kernel(
    const float* __restrict__ pf,
    const bf16* __restrict__ cwih_bf, const bf16* __restrict__ cwhh_bf,
    const float* __restrict__ cbih, const float* __restrict__ cbhh,
    const float* __restrict__ vol_w, const float* __restrict__ vol_b,
    const float* __restrict__ keysn, const float* __restrict__ mem_vals,
    const float* __restrict__ pfc_w, const float* __restrict__ pfc_b,
    const float* __restrict__ pfc_g, const float* __restrict__ pfc_beta,
    const float* __restrict__ hmono0,
    float* __restrict__ outs,
    float* __restrict__ hmono_out, float* __restrict__ scal_out) {
  __shared__ bf16 lat_s[BD];
  __shared__ bf16 hm_s[BD];
  __shared__ float gbuf[BB * G3];   // gate preacts; reused for scores+rel
  __shared__ float hnb[BD];         // saved hn preacts; reused for pfc out
  __shared__ float vol_s[BB];
  __shared__ float row_a[16];
  __shared__ float row_b[16];
  __shared__ float acc2[2];         // will, cost
  int tid = threadIdx.x;
  int wv = tid >> 5, lane = tid & 31;
  int g = lane >> 4, nlane = lane & 15;
  for (int i = tid; i < BD; i += 1024) hm_s[i] = (bf16)hmono0[i];
  if (tid == 0) { acc2[0] = 0.f; acc2[1] = 0.f; }
  __syncthreads();
  for (int t = 0; t < TT; ++t) {
    for (int i = tid; i < BD; i += 1024) {
      lat_s[i] = (bf16)pf[(size_t)t * BD + i];
      if (t + 1 < TT) __builtin_prefetch(pf + (size_t)(t + 1) * BD + i, 0, 0);
    }
    __syncthreads();
    for (int dep = 0; dep < DEPTH_C; ++dep) {
      if (tid < BB) {  // vol = sigmoid(lat . vol_w + vol_b + S_BIAS)
        float s = 0.f;
        for (int k = 0; k < DD; ++k) s += (float)lat_s[tid * DD + k] * vol_w[k];
        vol_s[tid] = sigf(s + vol_b[0] + S_BIAS_F);
      }
      __syncthreads();
      // gh = h_m @ cwhh^T + bhh  -> gbuf
      for (int nt = wv; nt < NT49; nt += 32) {
        v8f c = {};
#pragma unroll
        for (int kt = 0; kt < KT9; ++kt)
          c = wmma_bf16(afrag_bf16(hm_s, DD, BB, kt, lane), bfrag(cwhh_bf, nt, kt, lane), c);
        int n = nt * 16 + nlane;
        if (n < G3) {
#pragma unroll
          for (int r = 0; r < 8; ++r) {
            int m = g * 8 + r;
            if (m < BB) gbuf[m * G3 + n] = c[r] + cbhh[n];
          }
        }
      }
      __syncthreads();
      // gi = lat @ cwih^T + bih ; combine: r/z add in place, hn stashed
      for (int nt = wv; nt < NT49; nt += 32) {
        v8f c = {};
#pragma unroll
        for (int kt = 0; kt < KT9; ++kt)
          c = wmma_bf16(afrag_bf16(lat_s, DD, BB, kt, lane), bfrag(cwih_bf, nt, kt, lane), c);
        int n = nt * 16 + nlane;
        if (n < G3) {
#pragma unroll
          for (int r = 0; r < 8; ++r) {
            int m = g * 8 + r;
            if (m < BB) {
              float gv = c[r] + cbih[n];
              int o = m * G3 + n;
              if (n < 2 * DD) gbuf[o] += gv;
              else { hnb[m * DD + (n - 2 * DD)] = gbuf[o]; gbuf[o] = gv; }
            }
          }
        }
      }
      __syncthreads();
      if (tid == 0) {
        float vm = 0.f;
        for (int b = 0; b < BB; ++b) vm += vol_s[b];
        vm *= (1.f / BB);
        acc2[0] += vm;
        acc2[1] += vm * COST_STEP;
      }
      for (int idx = tid; idx < BD; idx += 1024) {
        int b = idx / DD, d = idx - b * DD;
        float rr = sigf(gbuf[b * G3 + d]);
        float zz = sigf(gbuf[b * G3 + DD + d]);
        float nn = tanhf(gbuf[b * G3 + 2 * DD + d] + rr * hnb[b * DD + d]);
        float hnew = (1.f - zz) * nn + zz * (float)hm_s[idx];
        hm_s[idx] = (bf16)hnew;
        lat_s[idx] = (bf16)((float)lat_s[idx] + hnew * vol_s[b]);
      }
      __syncthreads();
    }
    // ---- attention over memory (reuse gbuf: scores | rel) ----
    float* scores = gbuf;
    float* rel = gbuf + BB * MM;
    float* pfco = hnb;
    if (tid < BB) {
      float s = 0.f;
      for (int k = 0; k < DD; ++k) { float v = (float)lat_s[tid * DD + k]; s += v * v; }
      row_a[tid] = fmaxf(sqrtf(s), 1e-12f);
    }
    __syncthreads();
    for (int idx = tid; idx < BB * MM; idx += 1024) {
      int b = idx / MM, m = idx - b * MM;
      float s = 0.f;
      for (int k = 0; k < DD; ++k) s += (float)lat_s[b * DD + k] * keysn[m * DD + k];
      scores[idx] = s / row_a[b];
    }
    __syncthreads();
    if (tid < BB) {
      float mx = -1e30f;
      for (int m = 0; m < MM; ++m) mx = fmaxf(mx, scores[tid * MM + m]);
      float ssum = 0.f;
      for (int m = 0; m < MM; ++m) {
        float e = expf(scores[tid * MM + m] - mx);
        scores[tid * MM + m] = e; ssum += e;
      }
      float inv = 1.f / ssum;
      for (int m = 0; m < MM; ++m) scores[tid * MM + m] *= inv;
    }
    __syncthreads();
    for (int idx = tid; idx < BD; idx += 1024) {
      int b = idx / DD, d = idx - b * DD;
      float s = 0.f;
      for (int m = 0; m < MM; ++m) s += scores[b * MM + m] * mem_vals[m * DD + d];
      rel[idx] = s;
    }
    __syncthreads();
    // ---- pfc: cat(lat, rel) @ pfc_w^T + pfc_b ----
    for (int idx = tid; idx < BD; idx += 1024) {
      int b = idx / DD, d = idx - b * DD;
      float pre = pfc_b[d];
      const float* wrow = pfc_w + (size_t)d * (2 * DD);
      for (int j = 0; j < DD; ++j) pre += (float)lat_s[b * DD + j] * wrow[j];
      for (int j = 0; j < DD; ++j) pre += rel[b * DD + j] * wrow[DD + j];
      pfco[idx] = pre;
    }
    __syncthreads();
    if (tid < BB) {  // layernorm stats
      float mu = 0.f;
      for (int d = 0; d < DD; ++d) mu += pfco[tid * DD + d];
      mu /= DD;
      float var = 0.f;
      for (int d = 0; d < DD; ++d) { float dv = pfco[tid * DD + d] - mu; var += dv * dv; }
      var /= DD;
      row_a[tid] = mu;
      row_b[tid] = 1.f / sqrtf(var + 1e-5f);
    }
    __syncthreads();
    for (int idx = tid; idx < BD; idx += 1024) {
      int b = idx / DD, d = idx - b * DD;
      float xn = (pfco[idx] - row_a[b]) * row_b[b] * pfc_g[d] + pfc_beta[d];
      float ge = 0.5f * xn * (1.f + erff(xn * 0.70710678f));
      outs[(size_t)t * BD + idx] = ge;
    }
    __syncthreads();
  }
  for (int i = tid; i < BD; i += 1024) hmono_out[i] = (float)hm_s[i];
  if (tid == 0) {
    scal_out[0] = (float)DEPTH_C;                       // depth
    scal_out[1] = acc2[1];                              // cost
    scal_out[2] = acc2[0] / (float)(TT * DEPTH_C);      // avg_will
  }
}

// ---------------------------------------------------------------------------
// logits = LN(outs) @ l2n(soma_w)^T * gain  (M=B*T, N=256, K=260), WMMA bf16.
// ---------------------------------------------------------------------------
__global__ __launch_bounds__(256) void logits_kernel(const float* __restrict__ outs,
                                                     const bf16* __restrict__ soma_bf,
                                                     const float* __restrict__ on_g,
                                                     const float* __restrict__ on_beta,
                                                     const float* __restrict__ gain,
                                                     float* __restrict__ logits) {
  __shared__ float ys[16 * KPAD];
  __shared__ float mu[16];
  __shared__ float inv[16];
  int tid = threadIdx.x;
  int m0 = blockIdx.x * 16;
  for (int s = tid; s < 16 * KPAD; s += 256) {
    int r = s / KPAD, k = s - r * KPAD;
    int m = m0 + r;
    int b = m >> 13, tt = m & (TT - 1);   // m = b*T + t
    ys[s] = (k < DD) ? outs[((size_t)tt * BB + b) * DD + k] : 0.f;
  }
  __syncthreads();
  if (tid < 16) {
    float s = 0.f;
    for (int k = 0; k < DD; ++k) s += ys[tid * KPAD + k];
    float m = s / DD;
    float v = 0.f;
    for (int k = 0; k < DD; ++k) { float d = ys[tid * KPAD + k] - m; v += d * d; }
    mu[tid] = m;
    inv[tid] = 1.f / sqrtf(v / DD + 1e-5f);
  }
  __syncthreads();
  for (int s = tid; s < 16 * KPAD; s += 256) {
    int r = s / KPAD, k = s - r * KPAD;
    if (k < DD) ys[s] = (ys[s] - mu[r]) * inv[r] * on_g[k] + on_beta[k];
  }
  __syncthreads();
  float gn = gain[0];
  int wv = tid >> 5, lane = tid & 31;
  int g = lane >> 4, nlane = lane & 15;
  for (int nt = wv; nt < NT16; nt += 8) {
    v8f c = {};
#pragma unroll
    for (int kt = 0; kt < KT9; ++kt)
      c = wmma_bf16(afrag_f32(ys, KPAD, 16, kt, lane), bfrag(soma_bf, nt, kt, lane), c);
    int n = nt * 16 + nlane;
#pragma unroll
    for (int r = 0; r < 8; ++r) {
      int m = m0 + g * 8 + r;
      logits[(size_t)m * VV + n] = c[r] * gn;
    }
  }
}

// ---------------------------------------------------------------------------
extern "C" void kernel_launch(void* const* d_in, const int* in_sizes, int n_in,
                              void* d_out, int out_size, void* d_ws, size_t ws_size,
                              hipStream_t stream) {
  const int*   x        = (const int*)  d_in[0];
  const float* h_f      = (const float*)d_in[1];
  const float* h_mono   = (const float*)d_in[2];
  const float* soma_w   = (const float*)d_in[3];
  const float* gru_wih  = (const float*)d_in[4];
  const float* gru_whh  = (const float*)d_in[5];
  const float* gru_bih  = (const float*)d_in[6];
  const float* gru_bhh  = (const float*)d_in[7];
  const float* cell_wih = (const float*)d_in[8];
  const float* cell_whh = (const float*)d_in[9];
  const float* cell_bih = (const float*)d_in[10];
  const float* cell_bhh = (const float*)d_in[11];
  const float* vol_w    = (const float*)d_in[12];
  const float* vol_b    = (const float*)d_in[13];
  const float* pfc_w    = (const float*)d_in[14];
  const float* pfc_b    = (const float*)d_in[15];
  const float* pfc_g    = (const float*)d_in[16];
  const float* pfc_beta = (const float*)d_in[17];
  const float* on_g     = (const float*)d_in[18];
  const float* on_beta  = (const float*)d_in[19];
  const float* mem_keys = (const float*)d_in[20];
  const float* mem_vals = (const float*)d_in[21];
  const float* gain     = (const float*)d_in[22];
  (void)in_sizes; (void)n_in; (void)out_size; (void)ws_size;

  char* ws = (char*)d_ws;
  size_t off = 0;
  auto take = [&](size_t bytes) -> char* {
    char* p = ws + off;
    off = (off + bytes + 255) & ~(size_t)255;
    return p;
  };
  float* pf       = (float*)take((size_t)TT * BD * 4);           // (T,B,D) forward states
  float* outs     = (float*)take((size_t)TT * BD * 4);           // (T,B,D) tstep outputs
  float* gi       = (float*)take((size_t)BB * TT * G3 * 4);      // precomputed input gates
  bf16*  wih_bf   = (bf16*) take((size_t)NT49 * KT9 * 512 * 2);
  bf16*  whh_bf   = (bf16*) take((size_t)NT49 * KT9 * 512 * 2);
  bf16*  cwih_bf  = (bf16*) take((size_t)NT49 * KT9 * 512 * 2);
  bf16*  cwhh_bf  = (bf16*) take((size_t)NT49 * KT9 * 512 * 2);
  bf16*  soma_bf  = (bf16*) take((size_t)NT16 * KT9 * 512 * 2);
  float* keysn    = (float*)take((size_t)MM * DD * 4);
  float* somanorm = (float*)take((size_t)VV * 4);

  float* logits    = (float*)d_out;
  float* hf_out    = logits + (size_t)BB * TT * VV;
  float* hmono_out = hf_out + BD;
  float* scal_out  = hmono_out + BD;   // depth, cost, avg_will

  prep_norms_kernel<<<1, 1024, 0, stream>>>(mem_keys, soma_w, keysn, somanorm);
  swizzle_kernel<<<(NT49 * KT9 * 512 + 255) / 256, 256, 0, stream>>>(gru_wih,  nullptr,  G3, DD, NT49, wih_bf);
  swizzle_kernel<<<(NT49 * KT9 * 512 + 255) / 256, 256, 0, stream>>>(gru_whh,  nullptr,  G3, DD, NT49, whh_bf);
  swizzle_kernel<<<(NT49 * KT9 * 512 + 255) / 256, 256, 0, stream>>>(cell_wih, nullptr,  G3, DD, NT49, cwih_bf);
  swizzle_kernel<<<(NT49 * KT9 * 512 + 255) / 256, 256, 0, stream>>>(cell_whh, nullptr,  G3, DD, NT49, cwhh_bf);
  swizzle_kernel<<<(NT16 * KT9 * 512 + 255) / 256, 256, 0, stream>>>(soma_w,   somanorm, VV, DD, NT16, soma_bf);

  gi_kernel<<<BB * TT / 16, 256, 0, stream>>>(x, soma_w, wih_bf, gru_bih, gi);
  fgru_kernel<<<1, 1024, 0, stream>>>(gi, whh_bf, gru_bhh, h_f, pf, hf_out);
  tstep_kernel<<<1, 1024, 0, stream>>>(pf, cwih_bf, cwhh_bf, cell_bih, cell_bhh,
                                       vol_w, vol_b, keysn, mem_vals,
                                       pfc_w, pfc_b, pfc_g, pfc_beta,
                                       h_mono, outs, hmono_out, scal_out);
  logits_kernel<<<BB * TT / 16, 256, 0, stream>>>(outs, soma_bf, on_g, on_beta, gain, logits);
}
